// YOLO_12927851561274
// MI455X (gfx1250) — matrix operand order repaired and checked by reading, not compiled, not run on hardware
//
#include <hip/hip_runtime.h>

// ---- problem constants (from reference) ----
#define NB      32          // batch
#define NCLS    80          // class channels
#define NREG    64          // 4 * REG_MAX regression channels
#define A_TOTAL 21504       // 128^2 + 64^2 + 32^2
#define OFF0    0
#define OFF1    16384
#define OFF2    20480
#define TSTRIDE 68          // LDS row stride (words) for reg tile: 16B-aligned rows

typedef __attribute__((ext_vector_type(16))) _Float16 v16h;
typedef __attribute__((ext_vector_type(8)))  float    v8f;

// ---------------------------------------------------------------------------
// Kernel A: cls NCHW -> NHWC transpose via padded LDS tile (coalesced both ways)
// in : [B][C][HW]     out: base of [B][A_TOTAL][C] region, rows at anc_off+p
// ---------------------------------------------------------------------------
__global__ __launch_bounds__(256) void nchw_to_nhwc(
    const float* __restrict__ in, float* __restrict__ out,
    int C, int HW, int anc_off)
{
    __shared__ float tile[32][33];
    const int tx = threadIdx.x;          // 0..31
    const int ty = threadIdx.y;          // 0..7
    const int p0 = blockIdx.x * 32;      // HW is always a multiple of 32
    const int c0 = blockIdx.y * 32;
    const int b  = blockIdx.z;

    const float* inb = in + (size_t)b * C * HW;
    const int p = p0 + tx;
#pragma unroll
    for (int j = 0; j < 4; ++j) {
        int c = c0 + ty + 8 * j;
        if (c < C) tile[ty + 8 * j][tx] = inb[(size_t)c * HW + p]; // coalesced along p
    }
    __syncthreads();
    const int c = c0 + tx;
    if (c < C) {
        size_t obase = ((size_t)b * A_TOTAL + anc_off + p0) * C + c;
#pragma unroll
        for (int j = 0; j < 4; ++j) {
            int pl = ty + 8 * j;
            out[obase + (size_t)pl * C] = tile[tx][pl];           // coalesced along c
        }
    }
}

// ---------------------------------------------------------------------------
// Kernel B (fused): reg transpose + DFL box decode, one launch per level.
// Block stages a 128-position x 64-channel tile in LDS (tile[p][c]), then:
//   (1) streams NHWC pred_dist rows out with float4 stores (coalesced)
//   (2) each of the 8 waves decodes 16 anchors:
//       acc = E(16x64) x W(64x8) via two v_wmma_f32_16x16x32_f16
//       (cols 0..3 = sum e*k numerators, cols 4..7 = sum e denominators)
// Saves the 176MB pred_dist re-read of a separate decode kernel.
// ---------------------------------------------------------------------------
__global__ __launch_bounds__(256) void reg_fused(
    const float* __restrict__ in,        // [B][64][HW] this level's reg
    float* __restrict__ dist_out,        // [B][A_TOTAL][64] region base
    float* __restrict__ box_out,         // [B][A_TOTAL][4]
    float* __restrict__ anch_out,        // [A_TOTAL][2]
    float* __restrict__ str_out,         // [A_TOTAL]
    int HW, int anc_off, int lw, float strv)
{
    __shared__ float tile[128 * TSTRIDE];      // tile[p][c], 34816 B
    const int t  = threadIdx.x;
    const int p0 = blockIdx.x * 128;           // HW is a multiple of 128 at all levels
    const int b  = blockIdx.y;

    // ---- fill LDS: tile[p][c] = in[b][c][p0+p]  (coalesced global reads) ----
    const float* inb = in + (size_t)b * NREG * HW + p0;
#pragma unroll
    for (int j = 0; j < 32; ++j) {
        int flat = j * 256 + t;                // 0..8191
        int c  = flat >> 7;                    // 0..63
        int pl = flat & 127;                   // 0..127
        tile[pl * TSTRIDE + c] = inb[(size_t)c * HW + pl];
    }
    __syncthreads();

    // ---- (1) pred_dist NHWC stores: c-fast, float4, fully contiguous per wave ----
    {
        const int c4   = (t & 15) * 4;
        const int prow = t >> 4;
        size_t obase = ((size_t)b * A_TOTAL + anc_off + p0) * NREG;
#pragma unroll
        for (int j = 0; j < 8; ++j) {
            int pl = prow + j * 16;
            float4 v = *(const float4*)&tile[pl * TSTRIDE + c4];
            *(float4*)&dist_out[obase + (size_t)pl * NREG + c4] = v;
        }
    }

    // ---- (2) box decode: wave w owns anchors p0 + w*16 + 0..15 ----
    const int lane = t & 31;
    const int wave = t >> 5;
    const int m    = lane & 15;                // anchor row this lane feeds
    const int half = lane >> 4;                // A-layout K-half select
    const float* rowp = &tile[(wave * 16 + m) * TSTRIDE];

    // per-side maxima (cancel in softmax ratio; keep exp in (0,1] for f16)
    float mx[4];
#pragma unroll
    for (int s = 0; s < 4; ++s) {
        float4 a0 = *(const float4*)(rowp + 16 * s);
        float4 a1 = *(const float4*)(rowp + 16 * s + 4);
        float4 a2 = *(const float4*)(rowp + 16 * s + 8);
        float4 a3 = *(const float4*)(rowp + 16 * s + 12);
        float mm = fmaxf(fmaxf(fmaxf(a0.x, a0.y), fmaxf(a0.z, a0.w)),
                         fmaxf(fmaxf(a1.x, a1.y), fmaxf(a1.z, a1.w)));
        mm = fmaxf(mm, fmaxf(fmaxf(fmaxf(a2.x, a2.y), fmaxf(a2.z, a2.w)),
                             fmaxf(fmaxf(a3.x, a3.y), fmaxf(a3.z, a3.w))));
        mx[s] = mm;
    }

    // ISA 16-bit A 16x32 layout: lanes0-15 hold K={0..7,16..23}; lanes16-31 K={8..15,24..31}
    auto buildA = [&](int cb, float mxa, float mxb) -> v16h {
        const float* q = rowp + cb + half * 8;
        float4 x0 = *(const float4*)(q);
        float4 x1 = *(const float4*)(q + 4);
        float4 y0 = *(const float4*)(q + 16);
        float4 y1 = *(const float4*)(q + 20);
        float xs[8] = {x0.x, x0.y, x0.z, x0.w, x1.x, x1.y, x1.z, x1.w};
        float ys[8] = {y0.x, y0.y, y0.z, y0.w, y1.x, y1.y, y1.z, y1.w};
        v16h a;
#pragma unroll
        for (int i = 0; i < 8; ++i) {
            a[i]     = (_Float16)__expf(xs[i] - mxa);
            a[8 + i] = (_Float16)__expf(ys[i] - mxb);
        }
        return a;
    };
    // B operand 32x16: lane holds column n=lane&15 for K = {half*8..half*8+7, 16+half*8..16+half*8+7}.
    // column `side` -> proj weight (k%16); column `side+4` -> 1.0; else 0.
    auto buildB = [&](int sbase) -> v16h {
        v16h bm;
        const int  ncol = lane & 15;
#pragma unroll
        for (int e = 0; e < 16; ++e) {
            const int  k    = (e < 8) ? (half * 8 + e) : (16 + half * 8 + (e - 8));
            const int  side = sbase + (k >> 4);
            const bool isP  = (ncol == side);
            const bool isO  = (ncol == side + 4);
            bm[e] = isP ? (_Float16)(float)(k & 15) : (isO ? (_Float16)1.0f : (_Float16)0.0f);
        }
        return bm;
    };

    v8f acc = {};
    acc = __builtin_amdgcn_wmma_f32_16x16x32_f16(false, buildA(0,  mx[0], mx[1]),
                                                 false, buildB(0), (short)0, acc, false, false);
    acc = __builtin_amdgcn_wmma_f32_16x16x32_f16(false, buildA(32, mx[2], mx[3]),
                                                 false, buildB(2), (short)0, acc, false, false);

    // D layout: lane l, element v -> M = v + 8*(l>=16), N = l&15.
    // N=0..3 numerators, N=4..7 denominators; pull denominator from lane n+4 (same half).
    const int n     = lane & 15;
    const int srcl  = (lane & 16) | ((n + 4) & 15);
    const int wmask = (1 << lw) - 1;
#pragma unroll
    for (int v = 0; v < 8; ++v) {
        float den = __shfl(acc[v], srcl, 32);
        if (n < 4) {
            int mm = v + (half << 3);
            int pl = p0 + wave * 16 + mm;      // position within level
            int a  = anc_off + pl;             // global anchor index
            float ax = (float)(pl & wmask) + 0.5f;
            float ay = (float)(pl >> lw)   + 0.5f;
            float d  = acc[v] / den;
            float ac = (n & 1) ? ay : ax;
            box_out[((size_t)b * A_TOTAL + a) * 4 + n] = (n < 2) ? (ac - d) : (ac + d);
        }
    }

    // ---- anchors & strides (batch-independent) ----
    if (b == 0 && t < 128) {
        int pl = p0 + t;
        int a  = anc_off + pl;
        anch_out[2 * a + 0] = (float)(pl & wmask) + 0.5f;
        anch_out[2 * a + 1] = (float)(pl >> lw)   + 0.5f;
        str_out[a] = strv;
    }
}

// ---------------------------------------------------------------------------
extern "C" void kernel_launch(void* const* d_in, const int* in_sizes, int n_in,
                              void* d_out, int out_size, void* d_ws, size_t ws_size,
                              hipStream_t stream)
{
    const float* cls0 = (const float*)d_in[0];
    const float* reg0 = (const float*)d_in[1];
    const float* cls1 = (const float*)d_in[2];
    const float* reg1 = (const float*)d_in[3];
    const float* cls2 = (const float*)d_in[4];
    const float* reg2 = (const float*)d_in[5];

    float* out      = (float*)d_out;
    float* cls_out  = out;                                       // [B][A][80]
    float* box_out  = cls_out  + (size_t)NB * A_TOTAL * NCLS;    // [B][A][4]
    float* dist_out = box_out  + (size_t)NB * A_TOTAL * 4;       // [B][A][64]
    float* anch_out = dist_out + (size_t)NB * A_TOTAL * NREG;    // [A][2]
    float* str_out  = anch_out + (size_t)A_TOTAL * 2;            // [A]

    dim3 blkT(32, 8, 1);
    // cls transposes (NCHW -> NHWC)
    nchw_to_nhwc<<<dim3(16384 / 32, 3, NB), blkT, 0, stream>>>(cls0, cls_out, NCLS, 16384, OFF0);
    nchw_to_nhwc<<<dim3( 4096 / 32, 3, NB), blkT, 0, stream>>>(cls1, cls_out, NCLS,  4096, OFF1);
    nchw_to_nhwc<<<dim3( 1024 / 32, 3, NB), blkT, 0, stream>>>(cls2, cls_out, NCLS,  1024, OFF2);

    // fused reg transpose + WMMA box decode (+ anchors/strides), one launch per level
    reg_fused<<<dim3(16384 / 128, NB), 256, 0, stream>>>(reg0, dist_out, box_out, anch_out, str_out,
                                                         16384, OFF0, 7,  8.f);
    reg_fused<<<dim3( 4096 / 128, NB), 256, 0, stream>>>(reg1, dist_out, box_out, anch_out, str_out,
                                                          4096, OFF1, 6, 16.f);
    reg_fused<<<dim3( 1024 / 128, NB), 256, 0, stream>>>(reg2, dist_out, box_out, anch_out, str_out,
                                                          1024, OFF2, 5, 32.f);
}
